// GNN_GINEConv_86294482911409
// MI455X (gfx1250) — compile-verified, matrix operand order checked by reading, not run on hardware
//
#include <hip/hip_runtime.h>
#include <hip/hip_bf16.h>

typedef __attribute__((ext_vector_type(16))) _Float16 v16h;
typedef __attribute__((ext_vector_type(8)))  float    v8f;

#define GH   128          // hidden dim H
#define GL   2            // layers
#define GG   64           // graphs
#define GMID 64           // classifier mid

// ---------------------------------------------------------------------------
// zero fill
__global__ __launch_bounds__(256) void zero_kernel(float* __restrict__ p, long n) {
    long i = (long)blockIdx.x * 256 + threadIdx.x;
    long stride = (long)gridDim.x * 256;
    for (; i < n; i += stride) p[i] = 0.0f;
}

// ---------------------------------------------------------------------------
// h[i,c] = x[i]*enc_w[c] + enc_b[c]
__global__ __launch_bounds__(256) void encode_kernel(
    const float* __restrict__ x, const float* __restrict__ w,
    const float* __restrict__ b, float* __restrict__ h, int N_) {
    long tid = (long)blockIdx.x * 256 + threadIdx.x;
    int n = (int)(tid >> 7);
    int c = (int)(tid & 127);
    if (n >= N_) return;
    h[tid] = x[n] * w[c] + b[c];
}

// ---------------------------------------------------------------------------
// Pack fp32 weights [k][n] (128x128) into per-lane WMMA B-fragments (f16).
// packed idx = ((((l*2+g)*8 + nt)*4 + kt)*32 + lane)*16 + e
__global__ __launch_bounds__(256) void pack_weights_kernel(
    const float* __restrict__ w1, const float* __restrict__ w2,
    _Float16* __restrict__ packed) {
    int idx = blockIdx.x * 256 + threadIdx.x;
    if (idx >= GL * 2 * GH * GH) return;
    int e    = idx & 15;
    int lane = (idx >> 4) & 31;
    int kt   = (idx >> 9) & 3;
    int nt   = (idx >> 11) & 7;
    int g    = (idx >> 14) & 1;
    int l    = idx >> 15;
    int k = kt * 32 + ((e < 8) ? 0 : 16) + ((lane >> 4) * 8) + (e & 7);
    int n = nt * 16 + (lane & 15);
    const float* w = (g == 0 ? w1 : w2) + (long)l * GH * GH;
    packed[idx] = (_Float16)w[k * GH + n];
}

// ---------------------------------------------------------------------------
// Edge phase: msg = relu(h[src] + ea*edge_w + edge_b); atomic scatter-add to aggr[dst].
// One edge per 32 lanes, float4 per lane (coalesced 512B row).
__global__ __launch_bounds__(256) void edge_kernel(
    const float* __restrict__ h, const int* __restrict__ eidx,
    const float* __restrict__ eattr, const float* __restrict__ ew,
    const float* __restrict__ eb, float* __restrict__ aggr, int E_) {
    long tid = (long)blockIdx.x * 256 + threadIdx.x;
    long edge = tid >> 5;
    int lane = (int)(tid & 31);
    if (edge >= E_) return;
    int s = eidx[edge];
    int d = eidx[(long)E_ + edge];
    float ea = eattr[edge];
    int c0 = lane * 4;
    float4 hv = *(const float4*)(h + (long)s * GH + c0);
    float4 w4 = *(const float4*)(ew + c0);
    float4 b4 = *(const float4*)(eb + c0);
    float m0 = fmaxf(hv.x + ea * w4.x + b4.x, 0.0f);
    float m1 = fmaxf(hv.y + ea * w4.y + b4.y, 0.0f);
    float m2 = fmaxf(hv.z + ea * w4.z + b4.z, 0.0f);
    float m3 = fmaxf(hv.w + ea * w4.w + b4.w, 0.0f);
    float* ap = aggr + (long)d * GH + c0;
    atomicAdd(ap + 0, m0);
    atomicAdd(ap + 1, m1);
    atomicAdd(ap + 2, m2);
    atomicAdd(ap + 3, m3);
}

// ---------------------------------------------------------------------------
// Fused GINE MLP: z=(1+eps)h+aggr ; y1=relu(z@W1+b1) ; z2=y1@W2+b2 ; BN ; relu.
// Block = 128 threads (4 waves), each block one 16-row stripe, WMMA f16->f32.
__global__ __launch_bounds__(128) void gine_mlp_kernel(
    const float* __restrict__ h, const float* __restrict__ aggr,
    const float* __restrict__ eps_p, int layer,
    const _Float16* __restrict__ pw1, const float* __restrict__ b1,
    const _Float16* __restrict__ pw2, const float* __restrict__ b2,
    const float* __restrict__ gamma, const float* __restrict__ beta,
    const float* __restrict__ mean, const float* __restrict__ var,
    float* __restrict__ hout, int N_) {
    __shared__ _Float16 lds_y1[16 * GH];

    const int tid  = threadIdx.x;
    const int wave = tid >> 5;
    const int lane = tid & 31;
    const int row0 = blockIdx.x * 16;
    const float epsv = 1.0f + eps_p[layer];

    const int m     = lane & 15;            // A-fragment row within tile
    const int khalf = (lane >> 4) * 8;      // K sub-offset per half-wave
    const int rowA  = min(row0 + m, N_ - 1);
    const long rbase = (long)rowA * GH;

    // ---- A fragments for GEMM1, z computed on the fly --------------------
    v16h afrag[4];
    #pragma unroll
    for (int kt = 0; kt < 4; ++kt) {
        #pragma unroll
        for (int e = 0; e < 16; ++e) {
            int k = kt * 32 + ((e < 8) ? 0 : 16) + khalf + (e & 7);
            float z = epsv * h[rbase + k] + aggr[rbase + k];
            afrag[kt][e] = (_Float16)z;
        }
    }

    // ---- GEMM1: 2 column tiles per wave, bias+relu -> LDS (f16) ----------
    #pragma unroll
    for (int t = 0; t < 2; ++t) {
        int nt = wave * 2 + t;
        v8f c = {};
        #pragma unroll
        for (int kt = 0; kt < 4; ++kt) {
            v16h bfrag = *(const v16h*)(pw1 + (((long)(nt * 4 + kt) * 32 + lane) * 16));
            c = __builtin_amdgcn_wmma_f32_16x16x32_f16(
                    false, afrag[kt], false, bfrag, (short)0, c, false, false);
        }
        #pragma unroll
        for (int r = 0; r < 8; ++r) {
            int mm = r + (lane >> 4) * 8;
            int nn = nt * 16 + (lane & 15);
            float y = fmaxf(c[r] + b1[nn], 0.0f);
            lds_y1[mm * GH + nn] = (_Float16)y;
        }
    }
    __syncthreads();

    // ---- A fragments for GEMM2 from LDS ----------------------------------
    v16h afrag2[4];
    #pragma unroll
    for (int kt = 0; kt < 4; ++kt) {
        #pragma unroll
        for (int e = 0; e < 16; ++e) {
            int k = kt * 32 + ((e < 8) ? 0 : 16) + khalf + (e & 7);
            afrag2[kt][e] = lds_y1[m * GH + k];
        }
    }

    // ---- GEMM2 + bias + BatchNorm(eval) + relu ---------------------------
    #pragma unroll
    for (int t = 0; t < 2; ++t) {
        int nt = wave * 2 + t;
        v8f c = {};
        #pragma unroll
        for (int kt = 0; kt < 4; ++kt) {
            v16h bfrag = *(const v16h*)(pw2 + (((long)(nt * 4 + kt) * 32 + lane) * 16));
            c = __builtin_amdgcn_wmma_f32_16x16x32_f16(
                    false, afrag2[kt], false, bfrag, (short)0, c, false, false);
        }
        #pragma unroll
        for (int r = 0; r < 8; ++r) {
            int mm = r + (lane >> 4) * 8;
            int nn = nt * 16 + (lane & 15);
            if (row0 + mm < N_) {
                float z = c[r] + b2[nn];
                z = (z - mean[nn]) * rsqrtf(var[nn] + 1e-5f) * gamma[nn] + beta[nn];
                hout[(long)(row0 + mm) * GH + nn] = fmaxf(z, 0.0f);
            }
        }
    }
}

// ---------------------------------------------------------------------------
// Global mean pool accumulation: atomic segment-sum over graphs.
__global__ __launch_bounds__(256) void pool_kernel(
    const float* __restrict__ h, const int* __restrict__ batch,
    float* __restrict__ sums, float* __restrict__ cnts, int N_) {
    long tid = (long)blockIdx.x * 256 + threadIdx.x;
    int n = (int)(tid >> 5);
    int lane = (int)(tid & 31);
    if (n >= N_) return;
    int b = batch[n];
    int c0 = lane * 4;
    float4 hv = *(const float4*)(h + (long)n * GH + c0);
    float* sp = sums + (long)b * GH + c0;
    atomicAdd(sp + 0, hv.x);
    atomicAdd(sp + 1, hv.y);
    atomicAdd(sp + 2, hv.z);
    atomicAdd(sp + 3, hv.w);
    if (lane == 0) atomicAdd(cnts + b, 1.0f);
}

// ---------------------------------------------------------------------------
// Classifier: g = sums/cnt ; relu(g@W1+b1)@W2+b2 ; sigmoid. One block, 64 threads.
__global__ __launch_bounds__(64) void cls_kernel(
    const float* __restrict__ sums, const float* __restrict__ cnts,
    const float* __restrict__ w1, const float* __restrict__ b1,
    const float* __restrict__ w2, const float* __restrict__ b2,
    float* __restrict__ out) {
    __shared__ float g[GG * 129];   // stride 129 -> conflict-free banks
    int t = threadIdx.x;
    float cnt = fmaxf(cnts[t], 1.0f);
    for (int k = 0; k < GH; ++k) g[t * 129 + k] = sums[t * GH + k] / cnt;
    __syncthreads();
    float acc2 = b2[0];
    for (int m2 = 0; m2 < GMID; ++m2) {
        float a = b1[m2];
        for (int k = 0; k < GH; ++k) a += g[t * 129 + k] * w1[k * GMID + m2];
        acc2 += fmaxf(a, 0.0f) * w2[m2];
    }
    out[t] = 1.0f / (1.0f + expf(-acc2));
}

// ---------------------------------------------------------------------------
extern "C" void kernel_launch(void* const* d_in, const int* in_sizes, int n_in,
                              void* d_out, int out_size, void* d_ws, size_t ws_size,
                              hipStream_t stream) {
    const float* x        = (const float*)d_in[0];
    const int*   eidx     = (const int*)  d_in[1];
    const float* eattr    = (const float*)d_in[2];
    const int*   batch    = (const int*)  d_in[3];
    const float* enc_w    = (const float*)d_in[4];
    const float* enc_b    = (const float*)d_in[5];
    const float* edge_w   = (const float*)d_in[6];
    const float* edge_b   = (const float*)d_in[7];
    const float* eps      = (const float*)d_in[8];
    const float* mlp_w1   = (const float*)d_in[9];
    const float* mlp_b1   = (const float*)d_in[10];
    const float* mlp_w2   = (const float*)d_in[11];
    const float* mlp_b2   = (const float*)d_in[12];
    const float* bn_gamma = (const float*)d_in[13];
    const float* bn_beta  = (const float*)d_in[14];
    const float* bn_mean  = (const float*)d_in[15];
    const float* bn_var   = (const float*)d_in[16];
    const float* cls_w1   = (const float*)d_in[17];
    const float* cls_b1   = (const float*)d_in[18];
    const float* cls_w2   = (const float*)d_in[19];
    const float* cls_b2   = (const float*)d_in[20];

    const int N_ = in_sizes[0];
    const int E_ = in_sizes[2];

    // workspace layout
    float* hA   = (float*)d_ws;                 // N*H
    float* hB   = hA + (size_t)N_ * GH;         // N*H
    float* aggr = hB + (size_t)N_ * GH;         // N*H
    _Float16* packed = (_Float16*)(aggr + (size_t)N_ * GH);   // L*2*H*H halfs
    float* sums = (float*)(packed + (size_t)GL * 2 * GH * GH); // G*H
    float* cnts = sums + (size_t)GG * GH;                      // G

    // 1) pack weights into WMMA fragment layout
    {
        int tot = GL * 2 * GH * GH;
        pack_weights_kernel<<<(tot + 255) / 256, 256, 0, stream>>>(mlp_w1, mlp_w2, packed);
    }
    // 2) encoder
    {
        long tot = (long)N_ * GH;
        encode_kernel<<<(int)((tot + 255) / 256), 256, 0, stream>>>(x, enc_w, enc_b, hA, N_);
    }
    // 3) two GINE layers (ping-pong hA <-> hB)
    const float* hin = hA;
    float* hout = hB;
    for (int l = 0; l < GL; ++l) {
        long nh = (long)N_ * GH;
        zero_kernel<<<(int)((nh + 255) / 256), 256, 0, stream>>>(aggr, nh);
        long et = (long)E_ * 32;
        edge_kernel<<<(int)((et + 255) / 256), 256, 0, stream>>>(
            hin, eidx, eattr, edge_w, edge_b, aggr, E_);
        gine_mlp_kernel<<<(N_ + 15) / 16, 128, 0, stream>>>(
            hin, aggr, eps, l,
            packed + (size_t)(l * 2 + 0) * GH * GH, mlp_b1 + l * GH,
            packed + (size_t)(l * 2 + 1) * GH * GH, mlp_b2 + l * GH,
            bn_gamma + l * GH, bn_beta + l * GH, bn_mean + l * GH, bn_var + l * GH,
            hout, N_);
        const float* tmp = hin; hin = hout; hout = (float*)tmp;
    }
    // 4) pool + classifier
    {
        long nz = (long)GG * GH + GG;
        zero_kernel<<<(int)((nz + 255) / 256), 256, 0, stream>>>(sums, nz);
        long pt = (long)N_ * 32;
        pool_kernel<<<(int)((pt + 255) / 256), 256, 0, stream>>>(hin, batch, sums, cnts, N_);
        cls_kernel<<<1, 64, 0, stream>>>(sums, cnts, cls_w1, cls_b1, cls_w2, cls_b2,
                                         (float*)d_out);
    }
}
